// Rank1BayesianRNN_25469156065319
// MI455X (gfx1250) — matrix operand order repaired and checked by reading, not compiled, not run on hardware
//
#include <hip/hip_runtime.h>
#include <math.h>

// ---------------------------------------------------------------------------
// Rank1 Bayesian LSTM for MI455X (gfx1250).
//   * bf16 WMMA (v_wmma_f32_16x16x32_bf16), fp32 accumulate / fp32 cell state.
//   * Rank-1 alpha/gamma folded into per-ensemble bf16 weights, stored
//     K-MAJOR (pre-transposed) so weight tiles are dense K-runs.
//   * 32 MB folded weights -> fully L2 resident (192 MB L2); the recurrence
//     is WMMA-pipe bound, not HBM bound (23.3 TB/s barely touched).
//   * Weight tiles staged to LDS by the Tensor Data Mover (TDM) as a 3D
//     tile descriptor (32 K x 32 cols x 4 gate groups), tracked with
//     TENSORcnt; activations staged cooperatively (fp32->bf16 convert).
//   * Double-buffered LDS overlaps staging with the matrix pipe.
//   * One fused kernel per (timestep, layer): both GEMMs + gates + state.
// ---------------------------------------------------------------------------

typedef __bf16 bf16;
typedef __bf16 v16bf __attribute__((ext_vector_type(16)));
typedef __bf16 v8bf  __attribute__((ext_vector_type(8)));
typedef __bf16 v4bf  __attribute__((ext_vector_type(4)));
typedef float  v8f   __attribute__((ext_vector_type(8)));

typedef unsigned int u32x4 __attribute__((ext_vector_type(4)));
typedef int          i32x4 __attribute__((ext_vector_type(4)));
typedef int          i32x8 __attribute__((ext_vector_type(8)));

union AFrag { v16bf v; v8bf h[2]; };

#if defined(__has_builtin)
#  if __has_builtin(__builtin_amdgcn_tensor_load_to_lds) && \
      __has_builtin(__builtin_amdgcn_s_wait_tensorcnt)
#    define HAVE_TDM 1
#  endif
#endif
#ifndef HAVE_TDM
#  define HAVE_TDM 0
#endif

#define B_   256
#define T_   256
#define D_   512
#define H_   512
#define E_   4
#define L_   2
#define G4H  2048
#define CTX_ 256
#define HID_ 256
#define OUT_ 64
#define KFEAT 768   // H_ + CTX_

__device__ __forceinline__ float sigmoidf_(float x) {
  return 1.0f / (1.0f + __expf(-x));
}
__device__ __forceinline__ float tanh_fast(float x) {
  return 1.0f - 2.0f / (__expf(2.0f * x) + 1.0f);
}

__device__ __forceinline__ v8f wmma_bf16(v16bf a, v16bf b, v8f c) {
  return __builtin_amdgcn_wmma_f32_16x16x32_bf16(false, a, false, b,
                                                 (short)0, c, false, false);
}

#if HAVE_TDM
// Issue one TDM descriptor that pulls the 128-col x 32-K weight tile into
// LDS.  D# per CDNA5 ISA ch.8: 3D tile, X = 32 contiguous K bf16 elements,
// Y = 32 columns (stride tensor_dim0_stride = 512 elems), Z = 4 gate groups
// (stride tensor_dim1_stride = 512*512 elems).  The TDM writes LDS
// sequentially X->Y->Z, which is exactly our ldsB[c][k] layout.
__device__ __forceinline__ void tdm_load_b_tile(const bf16* gsrc,
                                                void* lds_dst) {
  const unsigned lds_off = (unsigned)(size_t)lds_dst;  // LDS byte offset
  const unsigned long long ga = (unsigned long long)(size_t)gsrc;
  u32x4 g0;
  g0[0] = 1u;                        // count=1, user descriptor
  g0[1] = lds_off;                   // lds_addr
  g0[2] = (unsigned)ga;              // global_addr[31:0]
  g0[3] = (unsigned)((ga >> 32) & 0x1FFFFFFu) | (2u << 30);  // [56:32]|type=2
  i32x8 g1;
  g1[0] = 0x00010000;                // data_size = 1 -> 2-byte elements
  g1[1] = (int)(512u << 16);         // tensor_dim0 = 512 (low 16 @ bits 63:48)
  g1[2] = (int)(32u << 16);          // tensor_dim0 hi = 0; tensor_dim1 = 32
  g1[3] = (int)(32u << 16);          // tensor_dim1 hi = 0; tile_dim0 = 32
  g1[4] = (int)((4u << 16) | 32u);   // tile_dim1 = 32; tile_dim2 = 4
  g1[5] = 512;                       // tensor_dim0_stride[31:0]
  g1[6] = 0;                         // stride0 hi; stride1 low16 (0x40000->0)
  g1[7] = 4;                         // tensor_dim1_stride[47:16] = 4 (0x40000)
  i32x4 g2;
  g2[0] = 4; g2[1] = 0; g2[2] = 0; g2[3] = 0;   // tensor_dim2 = 4
  i32x4 g3;
  g3[0] = 0; g3[1] = 0; g3[2] = 0; g3[3] = 0;
#if __clang_major__ >= 23
  i32x8 g4;
#pragma unroll
  for (int z = 0; z < 8; ++z) g4[z] = 0;
  __builtin_amdgcn_tensor_load_to_lds(g0, g1, g2, g3, g4, 0);
#else
  __builtin_amdgcn_tensor_load_to_lds(g0, g1, g2, g3, 0);
#endif
}
#endif  // HAVE_TDM

// ---------------------------------------------------------------------------
// Fold rank-1 factors into per-ensemble bf16 LSTM weights, K-MAJOR:
//   Wt[l][e][c][k] = W[l][k][c] * alpha[l][e][k]  * gamma[l][e][c]
//   Ut[l][e][c][k] = U[l][k][c] * ralpha[l][e][k] * rgamma[l][e][c]
// ---------------------------------------------------------------------------
__global__ __launch_bounds__(256) void k_conv_lstm(
    const float* __restrict__ lW, const float* __restrict__ lU,
    const float* __restrict__ la, const float* __restrict__ lg,
    const float* __restrict__ lra, const float* __restrict__ lrg,
    bf16* __restrict__ Wt, bf16* __restrict__ Ut)
{
  const size_t N = (size_t)L_ * E_ * G4H * D_;  // 8388608 per tensor
  for (size_t i = (size_t)blockIdx.x * 256 + threadIdx.x; i < 2 * N;
       i += (size_t)gridDim.x * 256) {
    const size_t j = (i < N) ? i : i - N;
    const int k = (int)(j & 511);
    const int c = (int)((j >> 9) & 2047);
    const int e = (int)((j >> 20) & 3);
    const int l = (int)((j >> 22) & 1);
    const size_t widx = ((size_t)l * D_ + k) * G4H + c;
    const size_t le = (size_t)l * E_ + e;
    if (i < N) {
      Wt[j] = (bf16)(lW[widx] * la[le * D_ + k] * lg[le * G4H + c]);
    } else {
      Ut[j] = (bf16)(lU[widx] * lra[le * H_ + k] * lrg[le * G4H + c]);
    }
  }
}

// Fold head rank-1 factors, K-major: hWt[e][c][k], oWt[e][c][k].
__global__ __launch_bounds__(256) void k_conv_head(
    const float* __restrict__ hW, const float* __restrict__ ha,
    const float* __restrict__ hg, const float* __restrict__ oW,
    const float* __restrict__ oa, const float* __restrict__ og,
    bf16* __restrict__ hWt, bf16* __restrict__ oWt)
{
  const size_t N1 = (size_t)E_ * HID_ * KFEAT;  // 786432
  const size_t N2 = (size_t)E_ * OUT_ * HID_;   // 65536
  for (size_t i = (size_t)blockIdx.x * 256 + threadIdx.x; i < N1 + N2;
       i += (size_t)gridDim.x * 256) {
    if (i < N1) {
      const int k = (int)(i % KFEAT);
      const int c = (int)((i / KFEAT) % HID_);
      const int e = (int)(i / ((size_t)KFEAT * HID_));
      hWt[i] = (bf16)(hW[(size_t)k * HID_ + c] * ha[(size_t)e * KFEAT + k] *
                      hg[(size_t)e * HID_ + c]);
    } else {
      const size_t j = i - N1;
      const int k = (int)(j & 255);
      const int c = (int)((j >> 8) & 63);
      const int e = (int)(j >> 14);
      oWt[j] = (bf16)(oW[(size_t)k * OUT_ + c] * oa[(size_t)e * HID_ + k] *
                      og[(size_t)e * OUT_ + c]);
    }
  }
}

// Zero-init LSTM state (h ping-pong bf16 buffers + fp32 cell state).
__global__ __launch_bounds__(256) void k_init(bf16* __restrict__ h,
                                              float* __restrict__ c)
{
  const int i = blockIdx.x * 256 + threadIdx.x;
  if (i < 2 * L_ * B_ * H_) h[i] = (bf16)0.0f;
  if (i < L_ * B_ * H_) c[i] = 0.0f;
}

// ---------------------------------------------------------------------------
// One fused (timestep, layer) LSTM step.
//   grid = (16 hcol-blocks of 32, 4 row-blocks of 64 == ensemble member)
//   block = 256 (8 waves); wave -> one 16x16 tile in each of the 4 gates, so
//   the gate math + state update run locally on the WMMA accumulators.
//   32 K-chunks (16 input GEMM + 16 recurrent GEMM), double-buffered LDS.
//   Weight tiles arrive via TDM (TENSORcnt); activations cooperatively.
// ---------------------------------------------------------------------------
template <int LAYER>
__global__ __launch_bounds__(256) void k_lstm_step(
    const float* __restrict__ seq, const int* __restrict__ slen,
    const float* __restrict__ lb,
    const bf16* __restrict__ Wt, const bf16* __restrict__ Ut,
    bf16* __restrict__ hbuf,   // [2][L][B][H] ping-pong
    float* __restrict__ cbuf,  // [L][B][H]
    int t)
{
  __shared__ bf16 ldsA[2][64 * 32];    // 64 rows x 32 K
  __shared__ bf16 ldsB[2][128 * 32];   // 128 cols (4 gates x 32) x 32 K, [c][k]

  const int hb = blockIdx.x;           // 0..15
  const int mb = blockIdx.y;           // 0..3 (== ensemble member e)
  const int e  = mb;
  const int tid   = threadIdx.x;
  const int wave  = tid >> 5;
  const int lane  = tid & 31;
  const int m     = wave >> 1;
  const int n     = wave & 1;
  const int lrow  = lane & 15;
  const int lhalf = lane >> 4;
  const int p  = t & 1;                // read parity
  const int pw = p ^ 1;                // write parity

  const bf16* Wbase = Wt + (size_t)(LAYER * E_ + e) * G4H * D_;  // K-major
  const bf16* Ubase = Ut + (size_t)(LAYER * E_ + e) * G4H * D_;
  const bf16* hprev = hbuf + (((size_t)p * L_ + LAYER) * B_) * H_;
  const bf16* h0new = hbuf + (((size_t)pw * L_ + 0) * B_) * H_;

  auto stage = [&](int s, int buf) {
    const int phase = s >> 4;
    const int kb = (s & 15) * 32;
    const bf16* wsrc = (phase == 0) ? Wbase : Ubase;
    // ---- B tile: 128 cols x 32 K of K-major weights ----
#if HAVE_TDM
    if (wave == 0)  // one TDM descriptor per workgroup (EXEC-independent)
      tdm_load_b_tile(&wsrc[(size_t)(hb * 32) * D_ + kb], &ldsB[buf][0]);
#else
#pragma unroll
    for (int it = 0; it < 2; ++it) {
      const int i = tid + it * 256;        // 512 x 16B
      const int c = i >> 2, part = i & 3;
      const int gc = (c >> 5) * 512 + hb * 32 + (c & 31);
      *(v8bf*)&ldsB[buf][c * 32 + part * 8] =
          *(const v8bf*)&wsrc[(size_t)gc * D_ + kb + part * 8];
    }
    if (s + 2 < 32) {  // keep the chunk-after-next slab warm
      const int kb2 = ((s + 2) & 15) * 32;
      const bf16* wsrc2 = (((s + 2) >> 4) == 0) ? Wbase : Ubase;
      const int c = tid >> 1;
      __builtin_prefetch(
          &wsrc2[(size_t)((c >> 5) * 512 + hb * 32 + (c & 31)) * D_ + kb2],
          0, 1);
    }
#endif
    // ---- A tile ----
    if (LAYER == 0 && phase == 0) {
      // fp32 sequence embeddings: float4 load, convert to bf16
#pragma unroll
      for (int it = 0; it < 2; ++it) {
        const int i = tid + it * 256;      // 512 x float4
        const int r = i >> 3, pp = i & 7;
        const int row = mb * 64 + r;
        const float4 x = *(const float4*)&seq[(size_t)row * (T_ * D_) +
                                              (size_t)t * D_ + kb + pp * 4];
        v4bf o;
        o[0] = (bf16)x.x; o[1] = (bf16)x.y; o[2] = (bf16)x.z; o[3] = (bf16)x.w;
        *(v4bf*)&ldsA[buf][r * 32 + pp * 4] = o;
      }
    } else {
      const bf16* abase = (phase == 0) ? h0new : hprev;
      const int r = tid >> 2, part = tid & 3;   // 256 x 16B
      const int row = mb * 64 + r;
      *(v8bf*)&ldsA[buf][r * 32 + part * 8] =
          *(const v8bf*)&abase[(size_t)row * H_ + kb + part * 8];
    }
  };

  v8f acc[4];
#pragma unroll
  for (int q = 0; q < 4; ++q)
#pragma unroll
    for (int k = 0; k < 8; ++k) acc[q][k] = 0.0f;

  stage(0, 0);
  for (int s = 0; s < 32; ++s) {
    const int cur = s & 1;
#if HAVE_TDM
    if (wave == 0) __builtin_amdgcn_s_wait_tensorcnt(0);  // buf `cur` landed
#endif
    __syncthreads();
    if (s + 1 < 32) stage(s + 1, cur ^ 1);
    // A fragment (contiguous K, matching B):
    //   lanes 0-15:  M=lane,    K 0-15
    //   lanes 16-31: M=lane-16, K 16-31
    AFrag af;
    const bf16* ap = &ldsA[cur][(m * 16 + lrow) * 32 + lhalf * 16];
    af.v = *(const v16bf*)(ap);
    const int bk = lhalf * 16;  // B: lanes 0-15 K 0-15, lanes 16-31 K 16-31
#pragma unroll
    for (int q = 0; q < 4; ++q) {
      v16bf bfm = *(const v16bf*)&ldsB[cur][(q * 32 + n * 16 + lrow) * 32 + bk];
      acc[q] = wmma_bf16(af.v, bfm, acc[q]);
    }
  }

  // --- fused gate nonlinearity + state update (accs hold i,f,g,o tiles) ---
  const int colh = hb * 32 + n * 16 + lrow;  // 0..511
  const size_t bbase = (size_t)(LAYER * E_ + e) * G4H + colh;
  const float bi  = lb[bbase + 0 * 512];
  const float bfv = lb[bbase + 1 * 512];
  const float bg  = lb[bbase + 2 * 512];
  const float bo  = lb[bbase + 3 * 512];

#pragma unroll
  for (int k = 0; k < 8; ++k) {
    const int row = mb * 64 + m * 16 + k + 8 * lhalf;  // C/D layout: M=k(+8)
    const size_t sidx = ((size_t)LAYER * B_ + row) * H_ + colh;
    const size_t hin  = (((size_t)p * L_ + LAYER) * B_ + row) * H_ + colh;
    const size_t hout = (((size_t)pw * L_ + LAYER) * B_ + row) * H_ + colh;
    const float cp = cbuf[sidx];
    const float iv = acc[0][k] + bi;
    const float fv = acc[1][k] + bfv;
    const float gv = acc[2][k] + bg;
    const float ov = acc[3][k] + bo;
    float cn = sigmoidf_(fv) * cp + sigmoidf_(iv) * tanh_fast(gv);
    float hn = sigmoidf_(ov) * tanh_fast(cn);
    if (t >= slen[row]) { cn = cp; hn = (float)hbuf[hin]; }  // masked step
    cbuf[sidx] = cn;
    hbuf[hout] = (bf16)hn;
  }
}

// ---------------------------------------------------------------------------
// Head layer 1: hidden = relu6( feat @ hidW' + b ),  feat = [h1 | ctx] (K=768)
// grid = (4 col-blocks of 64, 4 row-blocks of 64)
// ---------------------------------------------------------------------------
__global__ __launch_bounds__(256) void k_head_hidden(
    const bf16* __restrict__ hbuf, const float* __restrict__ ctx,
    const bf16* __restrict__ hWt, const float* __restrict__ hbias,
    float* __restrict__ hid, int fin)
{
  __shared__ bf16 ldsA[2][64 * 32];
  __shared__ bf16 ldsB[2][64 * 32];
  const int nb = blockIdx.x, mb = blockIdx.y, e = mb;
  const int tid = threadIdx.x, wave = tid >> 5, lane = tid & 31;
  const int m = wave & 3, np = wave >> 2;
  const int lrow = lane & 15, lhalf = lane >> 4;
  const bf16* hfin = hbuf + (((size_t)fin * L_ + 1) * B_) * H_;

  auto stage = [&](int s, int buf) {
    const int kb = s * 32;
    const int c = tid >> 2, part = tid & 3;   // 256 x 16B for B
    *(v8bf*)&ldsB[buf][c * 32 + part * 8] =
        *(const v8bf*)&hWt[((size_t)e * HID_ + nb * 64 + c) * KFEAT + kb +
                           part * 8];
    if (kb < H_) {
      const int r = tid >> 2;
      const int row = mb * 64 + r;
      *(v8bf*)&ldsA[buf][r * 32 + part * 8] =
          *(const v8bf*)&hfin[(size_t)row * H_ + kb + part * 8];
    } else {
#pragma unroll
      for (int it = 0; it < 2; ++it) {
        const int i = tid + it * 256;
        const int r = i >> 3, pp = i & 7;
        const int row = mb * 64 + r;
        const float4 x =
            *(const float4*)&ctx[(size_t)row * CTX_ + (kb - H_) + pp * 4];
        v4bf o;
        o[0] = (bf16)x.x; o[1] = (bf16)x.y; o[2] = (bf16)x.z; o[3] = (bf16)x.w;
        *(v4bf*)&ldsA[buf][r * 32 + pp * 4] = o;
      }
    }
  };

  v8f acc[2];
#pragma unroll
  for (int j = 0; j < 2; ++j)
#pragma unroll
    for (int k = 0; k < 8; ++k) acc[j][k] = 0.0f;

  stage(0, 0);
  for (int s = 0; s < KFEAT / 32; ++s) {
    const int cur = s & 1;
    __syncthreads();
    if (s + 1 < KFEAT / 32) stage(s + 1, cur ^ 1);
    AFrag af;
    const bf16* ap = &ldsA[cur][(m * 16 + lrow) * 32 + lhalf * 16];
    af.v = *(const v16bf*)(ap);
#pragma unroll
    for (int j = 0; j < 2; ++j) {
      const int c = (np * 2 + j) * 16 + lrow;
      v16bf bfm = *(const v16bf*)&ldsB[cur][c * 32 + lhalf * 16];
      acc[j] = wmma_bf16(af.v, bfm, acc[j]);
    }
  }
#pragma unroll
  for (int j = 0; j < 2; ++j)
#pragma unroll
    for (int k = 0; k < 8; ++k) {
      const int row = mb * 64 + m * 16 + k + 8 * lhalf;
      const int col = nb * 64 + (np * 2 + j) * 16 + lrow;
      float y = acc[j][k] + hbias[(size_t)e * HID_ + col];
      y = fminf(fmaxf(y, 0.0f), 6.0f);  // relu6
      hid[(size_t)row * HID_ + col] = y;
    }
}

// Head layer 2: out = hidden @ outW' + b   (M=256, N=64, K=256)
__global__ __launch_bounds__(256) void k_head_out(
    const float* __restrict__ hid, const bf16* __restrict__ oWt,
    const float* __restrict__ obias, float* __restrict__ out)
{
  __shared__ bf16 ldsA[2][64 * 32];
  __shared__ bf16 ldsB[2][64 * 32];
  const int mb = blockIdx.y, e = mb;
  const int tid = threadIdx.x, wave = tid >> 5, lane = tid & 31;
  const int m = wave & 3, np = wave >> 2;
  const int lrow = lane & 15, lhalf = lane >> 4;

  auto stage = [&](int s, int buf) {
    const int kb = s * 32;
    const int c = tid >> 2, part = tid & 3;  // c 0..63 used for B
    if (c < 64)
      *(v8bf*)&ldsB[buf][c * 32 + part * 8] =
          *(const v8bf*)&oWt[((size_t)e * OUT_ + c) * HID_ + kb + part * 8];
#pragma unroll
    for (int it = 0; it < 2; ++it) {
      const int i = tid + it * 256;
      const int r = i >> 3, pp = i & 7;
      const int row = mb * 64 + r;
      const float4 x = *(const float4*)&hid[(size_t)row * HID_ + kb + pp * 4];
      v4bf o;
      o[0] = (bf16)x.x; o[1] = (bf16)x.y; o[2] = (bf16)x.z; o[3] = (bf16)x.w;
      *(v4bf*)&ldsA[buf][r * 32 + pp * 4] = o;
    }
  };

  v8f acc[2];
#pragma unroll
  for (int j = 0; j < 2; ++j)
#pragma unroll
    for (int k = 0; k < 8; ++k) acc[j][k] = 0.0f;

  stage(0, 0);
  for (int s = 0; s < HID_ / 32; ++s) {
    const int cur = s & 1;
    __syncthreads();
    if (s + 1 < HID_ / 32) stage(s + 1, cur ^ 1);
    AFrag af;
    const bf16* ap = &ldsA[cur][(m * 16 + lrow) * 32 + lhalf * 16];
    af.v = *(const v16bf*)(ap);
#pragma unroll
    for (int j = 0; j < 2; ++j) {
      const int c = (np * 2 + j) * 16 + lrow;
      v16bf bfm = *(const v16bf*)&ldsB[cur][c * 32 + lhalf * 16];
      acc[j] = wmma_bf16(af.v, bfm, acc[j]);
    }
  }
#pragma unroll
  for (int j = 0; j < 2; ++j)
#pragma unroll
    for (int k = 0; k < 8; ++k) {
      const int row = mb * 64 + m * 16 + k + 8 * lhalf;
      const int col = (np * 2 + j) * 16 + lrow;  // < 64
      out[(size_t)row * OUT_ + col] = acc[j][k] + obias[(size_t)e * OUT_ + col];
    }
}

// ---------------------------------------------------------------------------
extern "C" void kernel_launch(void* const* d_in, const int* in_sizes, int n_in,
                              void* d_out, int out_size, void* d_ws,
                              size_t ws_size, hipStream_t stream)
{
  const float* seq  = (const float*)d_in[0];
  const float* ctx  = (const float*)d_in[1];
  const int*   slen = (const int*)d_in[2];
  const float* lW   = (const float*)d_in[3];
  const float* lU   = (const float*)d_in[4];
  const float* lb   = (const float*)d_in[5];
  const float* la   = (const float*)d_in[6];
  const float* lg   = (const float*)d_in[7];
  const float* lra  = (const float*)d_in[8];
  const float* lrg  = (const float*)d_in[9];
  const float* hW   = (const float*)d_in[10];
  const float* hb   = (const float*)d_in[11];
  const float* ha   = (const float*)d_in[12];
  const float* hg   = (const float*)d_in[13];
  const float* oW   = (const float*)d_in[14];
  const float* ob   = (const float*)d_in[15];
  const float* oa   = (const float*)d_in[16];
  const float* og   = (const float*)d_in[17];

  // Workspace carve (all regions fully rewritten every call -> deterministic)
  char* base = (char*)d_ws;
  size_t off = 0;
  auto carve = [&](size_t bytes) {
    void* r = base + off;
    off = (off + bytes + 255) & ~(size_t)255;
    return r;
  };
  bf16*  Wt     = (bf16*)carve((size_t)L_ * E_ * G4H * D_ * sizeof(bf16));
  bf16*  Ut     = (bf16*)carve((size_t)L_ * E_ * G4H * D_ * sizeof(bf16));
  bf16*  hWt    = (bf16*)carve((size_t)E_ * HID_ * KFEAT * sizeof(bf16));
  bf16*  oWt    = (bf16*)carve((size_t)E_ * OUT_ * HID_ * sizeof(bf16));
  bf16*  hbuf   = (bf16*)carve((size_t)2 * L_ * B_ * H_ * sizeof(bf16));
  float* cbuf   = (float*)carve((size_t)L_ * B_ * H_ * sizeof(float));
  float* hidwsp = (float*)carve((size_t)B_ * HID_ * sizeof(float));

  k_conv_lstm<<<16384, 256, 0, stream>>>(lW, lU, la, lg, lra, lrg, Wt, Ut);
  k_conv_head<<<3328, 256, 0, stream>>>(hW, ha, hg, oW, oa, og, hWt, oWt);
  k_init<<<2048, 256, 0, stream>>>(hbuf, cbuf);

  const dim3 g(16, 4);
  for (int t = 0; t < T_; ++t) {
    k_lstm_step<0><<<g, 256, 0, stream>>>(seq, slen, lb, Wt, Ut, hbuf, cbuf, t);
    k_lstm_step<1><<<g, 256, 0, stream>>>(seq, slen, lb, Wt, Ut, hbuf, cbuf, t);
  }

  // After step t the new h lives in parity (t+1)&1; final (t=255) -> parity 0.
  k_head_hidden<<<dim3(4, 4), 256, 0, stream>>>(hbuf, ctx, hWt, hb, hidwsp, 0);
  k_head_out<<<dim3(1, 4), 256, 0, stream>>>(hidwsp, oWt, ob, (float*)d_out);
}
